// LinearLoRA_34041910788892
// MI455X (gfx1250) — matrix operand order repaired and checked by reading, not compiled, not run on hardware
//
#include <hip/hip_runtime.h>

typedef __attribute__((ext_vector_type(2)))  float        v2f;
typedef __attribute__((ext_vector_type(8)))  float        v8f;
typedef __attribute__((ext_vector_type(16))) _Float16     v16h;
typedef __attribute__((ext_vector_type(4)))  unsigned int u32x4;
typedef __attribute__((ext_vector_type(8)))  int          i32x8;
typedef __attribute__((ext_vector_type(4)))  int          i32x4;

#define MDIM 16384   // B*S
#define NDIM 4096    // out features
#define KDIM 4096    // in features
#define RDIM 16      // LoRA rank

// ---------------------------------------------------------------------------
// Kernel A: low[M,16] (fp16) = x(fp16) @ lora_left_w^T   (r = 16)
// One wave per 16-row tile; x staged through LDS; v_wmma_f32_16x16x32_f16.
// ---------------------------------------------------------------------------
#define ABK   32
#define ALDSK 36

__global__ __launch_bounds__(256)
void lora_low_kernel(const float* __restrict__ x,
                     const _Float16* __restrict__ Lw,   // [16, KDIM] row-major
                     _Float16* __restrict__ low) {      // [MDIM, 16] row-major
    __shared__ float Xs[128][ALDSK];
    const int tid  = threadIdx.x;
    const int w    = tid >> 5;        // wave 0..7
    const int lane = tid & 31;
    const int l15  = lane & 15;
    const int g    = lane >> 4;       // lane half
    const int mbase = blockIdx.x * 128;

    v8f acc;
    #pragma unroll
    for (int e = 0; e < 8; ++e) acc[e] = 0.f;

    for (int k0 = 0; k0 < KDIM; k0 += ABK) {
        __syncthreads();
        #pragma unroll
        for (int i = 0; i < 4; ++i) {
            int e   = tid + i * 256;          // 0..1023 -> 128 rows x 8 float4
            int row = e >> 3;
            int c4  = (e & 7) << 2;
            *(float4*)&Xs[row][c4] =
                *(const float4*)(x + (size_t)(mbase + row) * KDIM + k0 + c4);
        }
        __syncthreads();

        v16h a;
        #pragma unroll
        for (int v = 0; v < 4; ++v) {
            int kb = 2 * v + g * 8;                  // K = 0..15
            a[2 * v]     = (_Float16)Xs[w * 16 + l15][kb];
            a[2 * v + 1] = (_Float16)Xs[w * 16 + l15][kb + 1];
        }
        #pragma unroll
        for (int v = 4; v < 8; ++v) {
            int kb = 2 * (v - 4) + 16 + g * 8;       // K = 16..31
            a[2 * v]     = (_Float16)Xs[w * 16 + l15][kb];
            a[2 * v + 1] = (_Float16)Xs[w * 16 + l15][kb + 1];
        }
        v16h bf;
        #pragma unroll
        for (int e = 0; e < 16; ++e) {
            bf[e] = Lw[(size_t)l15 * KDIM + k0 + e + g * 8 + (e & 8)];
        }
        acc = __builtin_amdgcn_wmma_f32_16x16x32_f16(
                  false, a, false, bf, (short)0, acc, false, false);
    }

    #pragma unroll
    for (int e = 0; e < 8; ++e) {
        int row = mbase + w * 16 + e + g * 8;
        low[(size_t)row * RDIM + l15] = (_Float16)acc[e];
    }
}

// ---------------------------------------------------------------------------
// Kernel B: out = x @ W^T + b + (low @ Rw^T) / 16
// 128x128 tile per block, 8 waves (2 in M x 4 in N), each wave 64x32 via
// 4x2 grid of 16x16 f32 accumulators; v_wmma_f32_16x16x4_f32 main path.
// K-tiles staged with the Tensor Data Mover (tensor_load_to_lds), double
// buffered, with hardware LDS padding (rows of 16+4 DWORDs, conflict-free).
// ---------------------------------------------------------------------------
#define BK   16
#define LDSK 20                      // 16 data + 4 pad DWORDs (TDM pad fields)
#define NKT  (KDIM / BK)             // 256 K-steps

// Issue one TDM 2D tile load: rows [row0, row0+128) x cols [k0, k0+BK) of a
// [tensor_rows x KDIM] fp32 row-major matrix into LDS at lds_addr, padded.
__device__ static inline void tdm_load_tile(const float* gbase, int row0, int k0,
                                            int tensor_rows, unsigned int lds_addr) {
    unsigned long long ga = (unsigned long long)(uintptr_t)gbase +
                            ((unsigned long long)row0 * KDIM + (unsigned long long)k0) * 4ull;
    u32x4 g0;
    g0[0] = 1u;                                            // count=1 (valid D#)
    g0[1] = lds_addr;                                      // LDS byte address
    g0[2] = (unsigned int)ga;                              // global_addr[31:0]
    g0[3] = ((unsigned int)(ga >> 32) & 0x1FFFFFFu)        // global_addr[56:32]
            | (2u << 30);                                  // type=2 ("image")
    i32x8 g1;
    g1[0] = (int)((2u << 16)        // data_size = 4 bytes
                | (1u << 20)        // pad_enable
                | (3u << 22)        // pad_interval: 16 DWORDs
                | (3u << 25));      // pad_amount:   4 DWORDs
    g1[1] = (int)((unsigned int)KDIM << 16);         // tensor_dim0[15:0] @ bits 63:48
    g1[2] = (int)((unsigned int)tensor_rows << 16);  // tensor_dim1[15:0] @ bits 95:80
    g1[3] = (int)((unsigned int)BK << 16);           // tile_dim0 @ bits 127:112
    g1[4] = 128;                                     // tile_dim1 @ bits 143:128
    g1[5] = KDIM;                                    // tensor_dim0_stride[31:0]
    g1[6] = 0;
    g1[7] = 0;
    i32x4 gz4;
    gz4[0] = 0; gz4[1] = 0; gz4[2] = 0; gz4[3] = 0;  // 2D tensor: groups 2/3 unused
    i32x8 gz8;
    #pragma unroll
    for (int e = 0; e < 8; ++e) gz8[e] = 0;
    // amdgpu-toolchain (clang-23) 6-arg form:
    // (u32x4 g0, i32x8 g1, i32x4, i32x4, i32x8, i32 cpol)
    __builtin_amdgcn_tensor_load_to_lds(g0, g1, gz4, gz4, gz8, 0);
}

__global__ __launch_bounds__(256)
void gemm_lora_kernel(const float* __restrict__ x,
                      const float* __restrict__ W,      // [NDIM, KDIM] row-major
                      const float* __restrict__ bias,   // [NDIM]
                      const _Float16* __restrict__ low, // [MDIM, 16]
                      const _Float16* __restrict__ Rw,  // [NDIM, 16] row-major
                      float* __restrict__ out) {        // [MDIM, NDIM]
    __shared__ float Xs[2][128][LDSK];
    __shared__ float Ws[2][128][LDSK];
    const int tid  = threadIdx.x;
    const int w    = tid >> 5;
    const int lane = tid & 31;
    const int l15  = lane & 15;
    const int g    = lane >> 4;
    const int wm   = w & 1;           // wave M index (0..1)
    const int wn   = w >> 1;          // wave N index (0..3)
    const int mbase = blockIdx.y * 128;
    const int nbase = blockIdx.x * 128;
    const bool issuer = (w == 0);     // wave 0 drives the TDM (uniform branch)

    const unsigned int xs_base  = (unsigned int)(uintptr_t)&Xs[0][0][0];
    const unsigned int ws_base  = (unsigned int)(uintptr_t)&Ws[0][0][0];
    const unsigned int bufbytes = 128u * LDSK * 4u;

    v8f acc[4][2];
    #pragma unroll
    for (int i = 0; i < 4; ++i)
        #pragma unroll
        for (int j = 0; j < 2; ++j)
            #pragma unroll
            for (int e = 0; e < 8; ++e) acc[i][j][e] = 0.f;

    if (issuer) {                     // prologue: fill buffer 0
        tdm_load_tile(x, mbase, 0, MDIM, xs_base);
        tdm_load_tile(W, nbase, 0, NDIM, ws_base);
    }

    for (int kt = 0; kt < NKT; ++kt) {
        const int cur = kt & 1;
        if (issuer) {
            if (kt + 1 < NKT) {       // prefetch next tile into other buffer
                const int nk = (kt + 1) * BK;
                tdm_load_tile(x, mbase, nk, MDIM, xs_base + (unsigned)(cur ^ 1) * bufbytes);
                tdm_load_tile(W, nbase, nk, NDIM, ws_base + (unsigned)(cur ^ 1) * bufbytes);
                __builtin_amdgcn_s_wait_tensorcnt(2);  // current buffer's 2 loads done
            } else {
                __builtin_amdgcn_s_wait_tensorcnt(0);
            }
        }
        __syncthreads();              // buffer `cur` visible to all waves

        #pragma unroll
        for (int kk = 0; kk < BK; kk += 4) {
            const int klo = kk + g * 2;   // f32 A 16x4: lane half picks K pair
            v2f a[4], bf[2];
            #pragma unroll
            for (int i = 0; i < 4; ++i) {
                a[i][0] = Xs[cur][wm * 64 + i * 16 + l15][klo];
                a[i][1] = Xs[cur][wm * 64 + i * 16 + l15][klo + 1];
            }
            #pragma unroll
            for (int j = 0; j < 2; ++j) {
                bf[j][0] = Ws[cur][wn * 32 + j * 16 + l15][klo];
                bf[j][1] = Ws[cur][wn * 32 + j * 16 + l15][klo + 1];
            }
            #pragma unroll
            for (int i = 0; i < 4; ++i)
                #pragma unroll
                for (int j = 0; j < 2; ++j)
                    acc[i][j] = __builtin_amdgcn_wmma_f32_16x16x4_f32(
                        false, a[i], false, bf[j], (short)0, acc[i][j],
                        false, false);
        }
        __syncthreads();              // all reads of `cur` done before it is refilled
    }

    // ---- Epilogue: bias + LoRA (r=16 -> zero-pad K to 32) --------------
    v16h br[2];
    float bv[2];
    #pragma unroll
    for (int j = 0; j < 2; ++j) {
        const int n0 = nbase + wn * 32 + j * 16;
        #pragma unroll
        for (int e = 0; e < 16; ++e) {
            // B[k][n] = Rw[n*16 + k]; elements 0-7 hold K=8*g+e, elements 8-15 (K>=16) -> 0
            _Float16 v = Rw[(size_t)(n0 + l15) * RDIM + (e & 7) + g * 8];
            br[j][e] = (e & 8) ? (_Float16)0.f : v;
        }
        bv[j] = bias[n0 + l15];
    }

    #pragma unroll
    for (int i = 0; i < 4; ++i) {
        const int m0 = mbase + wm * 64 + i * 16;
        v16h al;
        #pragma unroll
        for (int v = 0; v < 4; ++v) {
            int kb = 2 * v + g * 8;               // K = 0..15 (valid, r=16)
            al[2 * v]     = low[(size_t)(m0 + l15) * RDIM + kb];
            al[2 * v + 1] = low[(size_t)(m0 + l15) * RDIM + kb + 1];
        }
        #pragma unroll
        for (int e = 8; e < 16; ++e) al[e] = (_Float16)0.f;  // K = 16..31 -> 0

        #pragma unroll
        for (int j = 0; j < 2; ++j) {
            v8f z;
            #pragma unroll
            for (int e = 0; e < 8; ++e) z[e] = 0.f;
            v8f d = __builtin_amdgcn_wmma_f32_16x16x32_f16(
                        false, al, false, br[j], (short)0, z, false, false);
            const int n0 = nbase + wn * 32 + j * 16;
            #pragma unroll
            for (int e = 0; e < 8; ++e) {
                int row = m0 + e + g * 8;
                out[(size_t)row * NDIM + n0 + l15] =
                    acc[i][j][e] + bv[j] + d[e] * 0.0625f;
            }
        }
    }
}

// ---------------------------------------------------------------------------
extern "C" void kernel_launch(void* const* d_in, const int* in_sizes, int n_in,
                              void* d_out, int out_size, void* d_ws, size_t ws_size,
                              hipStream_t stream) {
    const float*    x  = (const float*)d_in[0];
    const float*    W  = (const float*)d_in[1];
    const float*    b  = (const float*)d_in[2];
    const _Float16* Lw = (const _Float16*)d_in[3];
    const _Float16* Rw = (const _Float16*)d_in[4];
    float*          out = (float*)d_out;
    _Float16*       low = (_Float16*)d_ws;   // MDIM*16 fp16 = 512 KB scratch

    lora_low_kernel<<<dim3(MDIM / 128), 256, 0, stream>>>(x, Lw, low);
    gemm_lora_kernel<<<dim3(NDIM / 128, MDIM / 128), 256, 0, stream>>>(
        x, W, b, low, Rw, out);
}
